// TransformerBlock_71691594105378
// MI455X (gfx1250) — matrix operand order repaired and checked by reading, not compiled, not run on hardware
//
#include <hip/hip_runtime.h>
#include <hip/hip_bf16.h>
#include <math.h>

// ---------------------------------------------------------------------------
// Transformer block for MI455X (gfx1250, wave32, WMMA + Tensor Data Mover).
// All GEMMs + attention use v_wmma_f32_16x16x32_bf16 (fp32 accumulate).
// GEMM A-tiles are staged into LDS by the TDM (tensor_load_to_lds) with
// hardware row padding; completion tracked with TENSORcnt.
// ---------------------------------------------------------------------------

#define D_MODEL 1024
#define N_HEADS 16
#define N_KV    8
#define HEAD_DIM 64
#define HIDDEN  4096
#define BATCH   4
#define SEQ     2048
#define ROWS    (BATCH * SEQ)          // 8192 token rows

typedef __attribute__((ext_vector_type(16))) __bf16 v16bf;
typedef __attribute__((ext_vector_type(8)))  float  v8f;
typedef __attribute__((ext_vector_type(4)))  unsigned v4u;
typedef __attribute__((ext_vector_type(8)))  int      v8i;
typedef __attribute__((ext_vector_type(4)))  int      v4i;

union FragBF {        // one WMMA 16-bit operand: 16 bf16 per lane = 32 bytes
    v16bf v;
    uint4 q[2];
    unsigned short u[16];
};

__device__ __forceinline__ unsigned short f2bf_bits(float f) {
    union { float f; unsigned u; } cv; cv.f = f;
    unsigned u = cv.u;
    unsigned rnd = (u >> 16) & 1u;          // round-to-nearest-even
    u += 0x7fffu + rnd;
    return (unsigned short)(u >> 16);
}
__device__ __forceinline__ float bf2f(unsigned short b) {
    union { unsigned u; float f; } cv; cv.u = ((unsigned)b) << 16;
    return cv.f;
}

// ---------------------------------------------------------------------------
// Generic bf16 GEMM:  out[M,N] = A[M,K] @ B[K,N]  (+ optional f32 residual)
// Block tile 128x128, K-step 32.  8 waves, each wave owns a 32x64 sub-tile
// (2 M-frags x 4 N-frags of 16x16).
//   A tile: staged by the Tensor Data Mover, 128x32 bf16 rows padded 64B->80B
//           so every fragment is a contiguous 16B LDS load.
//   B tile: staged transposed by the waves (needed for the B-operand layout).
// ---------------------------------------------------------------------------
#define TM 128
#define TN 128
#define TK 32
#define LDA_S 40      // 80B row stride (16 DW data + 4 DW TDM pad)
#define LDB_S 40

__global__ __launch_bounds__(256)
void gemm_bf16_wmma(const unsigned short* __restrict__ A,
                    const unsigned short* __restrict__ Bm,
                    int M, int N, int K,
                    unsigned short* __restrict__ outB,
                    float* __restrict__ outF,
                    const float* __restrict__ res)
{
    __shared__ __align__(16) unsigned short Asm[TM * LDA_S];
    __shared__ __align__(16) unsigned short Bsm[TN * LDB_S];

    const int tid  = threadIdx.x;
    const int lane = tid & 31;
    const int wave = tid >> 5;
    const int l16  = lane & 15;
    const bool hi  = lane >= 16;
    const int wm   = wave & 3;        // 4 waves along M (4*32 = 128)
    const int wn   = wave >> 2;       // 2 waves along N (2*64 = 128)
    const int m0   = blockIdx.y * TM;
    const int n0   = blockIdx.x * TN;

    // LDS byte address of the A staging buffer (flat LDS addr low 32 bits)
    const unsigned lds_a = (unsigned)(unsigned long long)(&Asm[0]);
    const unsigned long long gaBase =
        (unsigned long long)(A + (size_t)m0 * K);   // byte address (2B elems)

    v8f acc[2][4];
    #pragma unroll
    for (int i = 0; i < 2; ++i)
        #pragma unroll
        for (int j = 0; j < 4; ++j)
            acc[i][j] = (v8f){0.f,0.f,0.f,0.f,0.f,0.f,0.f,0.f};

    for (int k0 = 0; k0 < K; k0 += TK) {
        // ---- A tile via TDM: 2-D tile 32(elem) x 128(rows), data_size=2B,
        //      pad_interval = 16 DWORDs (code 3), pad_amount = 4 DWORDs
        //      (code 3) -> LDS row stride 40 elements.  Wave 0 issues.
        if (wave == 0) {
            unsigned long long ga = gaBase + (unsigned long long)k0 * 2ull;
            v4u g0 = (v4u){
                1u,                                   // count=1 valid D#
                lds_a,                                // lds_addr
                (unsigned)ga,                         // global_addr[31:0]
                (unsigned)((ga >> 32) & 0x1FFFFFFull) // global_addr[56:32]
                    | (2u << 30)                      // type = image
            };
            unsigned td0 = (unsigned)K, td1 = (unsigned)M;
            unsigned long long st0 = (unsigned long long)K;
            v8i g1 = (v8i){
                (int)((1u << 16) |                    // data_size = 2 bytes
                      (1u << 20) |                    // pad_enable
                      (3u << 22) |                    // pad_interval: 16 DW
                      (3u << 25)),                    // pad_amount: 4 DW
                (int)((td0 & 0xFFFFu) << 16),                    // dim0 lo
                (int)(((td0 >> 16) & 0xFFFFu) | ((td1 & 0xFFFFu) << 16)),
                (int)(((td1 >> 16) & 0xFFFFu) | (32u << 16)),    // tile_dim0
                (int)128,                                        // tile_dim1
                (int)(unsigned)(st0 & 0xFFFFFFFFull),            // stride0 lo
                (int)(unsigned)((st0 >> 32) & 0xFFFFull),        // stride0 hi
                0
            };
            v4i z4 = (v4i){0,0,0,0};
            v8i z8 = (v8i){0,0,0,0,0,0,0,0};
            __builtin_amdgcn_tensor_load_to_lds(g0, g1, z4, z4, z8, 0);
        }

        // ---- stage B tile transposed (32 x 128 -> Bsm[col][k]); all global
        //      loads issued before any LDS store so they overlap the TDM.
        uint4 vb[2];
        #pragma unroll
        for (int it = 0; it < 2; ++it) {
            int idx = tid + it * 256;
            int kr = idx >> 4, seg = idx & 15;
            vb[it] = *((const uint4*)(Bm + (size_t)(k0 + kr) * N + n0) + seg);
        }
        #pragma unroll
        for (int it = 0; it < 2; ++it) {
            int idx = tid + it * 256;
            int kr = idx >> 4, seg = idx & 15;
            unsigned short us[8];
            __builtin_memcpy(us, &vb[it], 16);
            int nc = seg * 8;
            #pragma unroll
            for (int j = 0; j < 8; ++j) Bsm[(nc + j) * LDB_S + kr] = us[j];
        }
        if (k0 + TK < K)    // hint next B tile into cache (global_prefetch_b8)
            __builtin_prefetch(Bm + (size_t)(k0 + TK + (tid & 31)) * N + n0 + ((tid >> 5) << 4), 0, 0);

        if (wave == 0) __builtin_amdgcn_s_wait_tensorcnt(0);
        __syncthreads();

        // ---- fragments + WMMA
        FragBF a[2];
        #pragma unroll
        for (int am = 0; am < 2; ++am) {
            int r  = wm * 32 + am * 16 + l16;
            int ko = hi ? 8 : 0;               // A layout: lanes>=16 start at K=8
            a[am].q[0] = *((const uint4*)&Asm[r * LDA_S + ko]);
            a[am].q[1] = *((const uint4*)&Asm[r * LDA_S + ko + 16]);
        }
        #pragma unroll
        for (int bn = 0; bn < 4; ++bn) {
            FragBF b;
            int c  = wn * 64 + bn * 16 + l16;
            int ko = hi ? 8 : 0;               // B layout: lanes>=16 start at K=8
            b.q[0] = *((const uint4*)&Bsm[c * LDB_S + ko]);
            b.q[1] = *((const uint4*)&Bsm[c * LDB_S + ko + 16]);
            acc[0][bn] = __builtin_amdgcn_wmma_f32_16x16x32_bf16(
                false, a[0].v, false, b.v, (short)0, acc[0][bn], false, false);
            acc[1][bn] = __builtin_amdgcn_wmma_f32_16x16x32_bf16(
                false, a[1].v, false, b.v, (short)0, acc[1][bn], false, false);
        }
        __syncthreads();
    }

    // ---- epilogue (C layout: lanes<16 rows 0..7, lanes>=16 rows 8..15)
    #pragma unroll
    for (int am = 0; am < 2; ++am)
        #pragma unroll
        for (int bn = 0; bn < 4; ++bn)
            #pragma unroll
            for (int r = 0; r < 8; ++r) {
                int row = m0 + wm * 32 + am * 16 + r + (hi ? 8 : 0);
                int col = n0 + wn * 64 + bn * 16 + l16;
                size_t o = (size_t)row * N + col;
                float v = acc[am][bn][r];
                if (res)  v += res[o];
                if (outF) outF[o] = v;
                if (outB) outB[o] = f2bf_bits(v);
            }
}

// ---------------------------------------------------------------------------
// Flash attention: causal GQA.  One block per (q-tile of 64 rows, b*h).
// 4 waves x 16 q-rows. S=Q*K^T and O+=P*V via bf16 WMMA; online softmax
// with per-row m,l carried in lane registers (lanes 0..15 own rows).
// ---------------------------------------------------------------------------
__global__ __launch_bounds__(128)
void flash_attn_wmma(const unsigned short* __restrict__ qkv,  // [ROWS, 2048]
                     unsigned short* __restrict__ ctx)        // [ROWS, 1024]
{
    __shared__ __align__(16) unsigned short Ksm[64 * 72];      // [key][d]
    __shared__ __align__(16) unsigned short Vsm[64 * 72];      // [d][key] (T)
    __shared__ __align__(16) unsigned short Psm[4][16 * 72];   // per-wave P
    __shared__ float  Ssm[4][16 * 64];
    __shared__ float  alpha_s[4][16];
    __shared__ float  lrow_s[4][16];

    const int tid  = threadIdx.x;
    const int lane = tid & 31;
    const int w    = tid >> 5;
    const int l16  = lane & 15;
    const bool hi  = lane >= 16;
    const int qt   = blockIdx.x;              // q tile (64 rows)
    const int bh   = blockIdx.y;
    const int b    = bh >> 4, h = bh & 15, kvh = h >> 1;
    const int qoff = h * HEAD_DIM;
    const int koff = D_MODEL + kvh * HEAD_DIM;
    const int voff = D_MODEL + N_KV * HEAD_DIM + kvh * HEAD_DIM;
    const size_t brow = (size_t)b * SEQ;

    // ---- Q fragments (held for the whole kernel): A layout, d split 2x32
    FragBF qa[2];
    {
        int trow = qt * 64 + w * 16 + l16;
        const unsigned short* qp = qkv + (brow + trow) * 2048 + qoff;
        int ko = hi ? 8 : 0;
        #pragma unroll
        for (int c = 0; c < 2; ++c) {
            qa[c].q[0] = *((const uint4*)(qp + c * 32 + ko));
            qa[c].q[1] = *((const uint4*)(qp + c * 32 + ko + 16));
        }
    }

    v8f oacc[4];
    #pragma unroll
    for (int i = 0; i < 4; ++i) oacc[i] = (v8f){0.f,0.f,0.f,0.f,0.f,0.f,0.f,0.f};
    float m_i = -3.0e38f, l_i = 0.f;          // valid in lanes 0..15

    for (int kt = 0; kt <= qt; ++kt) {
        // ---- stage K (row-major) and V (transposed)
        #pragma unroll
        for (int it = 0; it < 4; ++it) {
            int idx = tid + it * 128;
            int key = idx >> 3, seg = idx & 7;
            const unsigned short* kp = qkv + (brow + kt * 64 + key) * 2048;
            uint4 kv4 = *((const uint4*)(kp + koff) + seg);
            *((uint4*)&Ksm[key * 72 + seg * 8]) = kv4;
            uint4 vv4 = *((const uint4*)(kp + voff) + seg);
            unsigned short us[8];
            __builtin_memcpy(us, &vv4, 16);
            int d0 = seg * 8;
            #pragma unroll
            for (int j = 0; j < 8; ++j) Vsm[(d0 + j) * 72 + key] = us[j];
        }
        __syncthreads();

        // ---- S = Q @ K^T   (B-frag cols = keys, k-dim = d)
        v8f s[4];
        #pragma unroll
        for (int bn = 0; bn < 4; ++bn) {
            v8f z = (v8f){0.f,0.f,0.f,0.f,0.f,0.f,0.f,0.f};
            int key = bn * 16 + l16;
            int ko2 = hi ? 8 : 0;
            FragBF kb;
            kb.q[0] = *((const uint4*)&Ksm[key * 72 + ko2]);
            kb.q[1] = *((const uint4*)&Ksm[key * 72 + ko2 + 16]);
            z = __builtin_amdgcn_wmma_f32_16x16x32_bf16(
                false, qa[0].v, false, kb.v, (short)0, z, false, false);
            kb.q[0] = *((const uint4*)&Ksm[key * 72 + 32 + ko2]);
            kb.q[1] = *((const uint4*)&Ksm[key * 72 + 32 + ko2 + 16]);
            s[bn] = __builtin_amdgcn_wmma_f32_16x16x32_bf16(
                false, qa[1].v, false, kb.v, (short)0, z, false, false);
        }
        #pragma unroll
        for (int bn = 0; bn < 4; ++bn)
            #pragma unroll
            for (int r = 0; r < 8; ++r)
                Ssm[w][(r + (hi ? 8 : 0)) * 64 + bn * 16 + l16] = s[bn][r];
        __syncthreads();

        // ---- online softmax: lane l16 owns row l16 of this wave
        if (lane < 16) {
            float* srow = &Ssm[w][l16 * 64];
            int tg = qt * 64 + w * 16 + l16;
            float mx = m_i;
            #pragma unroll 8
            for (int c = 0; c < 64; ++c) {
                int kg = kt * 64 + c;
                float sv = srow[c] * 0.125f;           // 1/sqrt(64)
                sv = (kg <= tg) ? sv : -3.0e38f;       // causal mask
                srow[c] = sv;
                mx = fmaxf(mx, sv);
            }
            float alpha = __expf(m_i - mx);
            float sum = 0.f;
            unsigned short* prow = &Psm[w][l16 * 72];
            #pragma unroll 8
            for (int c = 0; c < 64; ++c) {
                float p = __expf(srow[c] - mx);
                sum += p;
                prow[c] = f2bf_bits(p);
            }
            l_i = l_i * alpha + sum;
            m_i = mx;
            alpha_s[w][l16] = alpha;
        }
        __syncthreads();

        // ---- rescale O, then O += P @ V
        #pragma unroll
        for (int dn = 0; dn < 4; ++dn)
            #pragma unroll
            for (int r = 0; r < 8; ++r)
                oacc[dn][r] *= alpha_s[w][r + (hi ? 8 : 0)];

        #pragma unroll
        for (int c = 0; c < 2; ++c) {
            FragBF pa;
            int ko3 = hi ? 8 : 0;
            pa.q[0] = *((const uint4*)&Psm[w][l16 * 72 + c * 32 + ko3]);
            pa.q[1] = *((const uint4*)&Psm[w][l16 * 72 + c * 32 + ko3 + 16]);
            #pragma unroll
            for (int dn = 0; dn < 4; ++dn) {
                FragBF vbf;
                int d   = dn * 16 + l16;
                int ko2 = hi ? 8 : 0;
                vbf.q[0] = *((const uint4*)&Vsm[d * 72 + c * 32 + ko2]);
                vbf.q[1] = *((const uint4*)&Vsm[d * 72 + c * 32 + ko2 + 16]);
                oacc[dn] = __builtin_amdgcn_wmma_f32_16x16x32_bf16(
                    false, pa.v, false, vbf.v, (short)0, oacc[dn], false, false);
            }
        }
        __syncthreads();
    }

    if (lane < 16) lrow_s[w][l16] = l_i;
    __syncthreads();

    // ---- write context, layout [b, t, h, d] so WO-GEMM reads rows directly
    #pragma unroll
    for (int dn = 0; dn < 4; ++dn)
        #pragma unroll
        for (int r = 0; r < 8; ++r) {
            int row  = r + (hi ? 8 : 0);
            float li = 1.0f / lrow_s[w][row];
            int trow = qt * 64 + w * 16 + row;
            ctx[(brow + trow) * D_MODEL + h * HEAD_DIM + dn * 16 + l16] =
                f2bf_bits(oacc[dn][r] * li);
        }
}

// ---------------------------------------------------------------------------
// Elementwise kernels
// ---------------------------------------------------------------------------
__global__ __launch_bounds__(256)
void rmsnorm_bf16(const float* __restrict__ x, const float* __restrict__ g,
                  unsigned short* __restrict__ out)
{
    __shared__ float red[8];
    const int row = blockIdx.x;
    const float* xr = x + (size_t)row * D_MODEL;
    float ss = 0.f;
    #pragma unroll
    for (int i = 0; i < 4; ++i) {
        float v = xr[threadIdx.x + i * 256];
        ss += v * v;
    }
    #pragma unroll
    for (int off = 16; off > 0; off >>= 1) ss += __shfl_xor(ss, off, 32);
    if ((threadIdx.x & 31) == 0) red[threadIdx.x >> 5] = ss;
    __syncthreads();
    float total = 0.f;
    #pragma unroll
    for (int i = 0; i < 8; ++i) total += red[i];
    float rs = rsqrtf(total / (float)D_MODEL + 1e-6f);
    #pragma unroll
    for (int i = 0; i < 4; ++i) {
        int c = threadIdx.x + i * 256;
        out[(size_t)row * D_MODEL + c] = f2bf_bits(xr[c] * rs * g[c]);
    }
}

// fp32 -> bf16 with column-concat into a wider destination
__global__ __launch_bounds__(256)
void convert_concat(const float* __restrict__ src, unsigned short* __restrict__ dst,
                    int cols, int ldd, int coff, int total)
{
    int i = blockIdx.x * 256 + threadIdx.x;
    if (i >= total) return;
    int r = i / cols, c = i % cols;
    dst[(size_t)r * ldd + coff + c] = f2bf_bits(src[i]);
}

// half-split RoPE applied in-place to Q (16 heads) and K (8 heads) of qkv
__global__ __launch_bounds__(256)
void rope_qk(unsigned short* __restrict__ qkv)
{
    const int total = ROWS * (N_HEADS + N_KV) * (HEAD_DIM / 2);
    int i = blockIdx.x * 256 + threadIdx.x;
    if (i >= total) return;
    int row = i / ((N_HEADS + N_KV) * 32);
    int rem = i % ((N_HEADS + N_KV) * 32);
    int hh = rem / 32, j = rem % 32;
    int col = (hh < N_HEADS) ? hh * HEAD_DIM : D_MODEL + (hh - N_HEADS) * HEAD_DIM;
    int t = row % SEQ;
    float inv = __expf(-((float)(2 * j) / (float)HEAD_DIM) * 9.2103403719761836f);
    float ang = (float)t * inv;
    float cs = __cosf(ang), sn = __sinf(ang);
    unsigned short* base = qkv + (size_t)row * 2048 + col;
    float x1 = bf2f(base[j]), x2 = bf2f(base[j + 32]);
    base[j]      = f2bf_bits(x1 * cs - x2 * sn);
    base[j + 32] = f2bf_bits(x2 * cs + x1 * sn);
}

// h = silu(a) * b  where a = g12[:, :4096], b = g12[:, 4096:]
__global__ __launch_bounds__(256)
void silu_mul(const unsigned short* __restrict__ g12, unsigned short* __restrict__ out)
{
    const int total = ROWS * HIDDEN;
    int i = blockIdx.x * 256 + threadIdx.x;
    if (i >= total) return;
    int r = i / HIDDEN, c = i % HIDDEN;
    float a = bf2f(g12[(size_t)r * (2 * HIDDEN) + c]);
    float bv = bf2f(g12[(size_t)r * (2 * HIDDEN) + HIDDEN + c]);
    float s = a / (1.0f + __expf(-a));
    out[i] = f2bf_bits(s * bv);
}

// ---------------------------------------------------------------------------
// Host-side orchestration
// ---------------------------------------------------------------------------
extern "C" void kernel_launch(void* const* d_in, const int* in_sizes, int n_in,
                              void* d_out, int out_size, void* d_ws, size_t ws_size,
                              hipStream_t stream)
{
    const float* x  = (const float*)d_in[0];
    const float* g1 = (const float*)d_in[1];
    const float* g2 = (const float*)d_in[2];
    const float* wq = (const float*)d_in[3];
    const float* wk = (const float*)d_in[4];
    const float* wv = (const float*)d_in[5];
    const float* wo = (const float*)d_in[6];
    const float* w1 = (const float*)d_in[7];
    const float* w2 = (const float*)d_in[8];
    const float* w3 = (const float*)d_in[9];
    float* out = (float*)d_out;

    // workspace carve-up (bf16 buffers stored as ushort bits)
    char* ws = (char*)d_ws;
    size_t off = 0;
    auto alloc = [&](size_t bytes) { char* p = ws + off; off += (bytes + 255) & ~(size_t)255; return p; };
    unsigned short* wqkv_b = (unsigned short*)alloc((size_t)D_MODEL * 2048 * 2);
    unsigned short* wo_b   = (unsigned short*)alloc((size_t)D_MODEL * D_MODEL * 2);
    unsigned short* w12_b  = (unsigned short*)alloc((size_t)D_MODEL * 2 * HIDDEN * 2);
    unsigned short* w3_b   = (unsigned short*)alloc((size_t)HIDDEN * D_MODEL * 2);
    unsigned short* h_b    = (unsigned short*)alloc((size_t)ROWS * D_MODEL * 2);
    unsigned short* qkv_b  = (unsigned short*)alloc((size_t)ROWS * 2048 * 2);
    unsigned short* ctx_b  = (unsigned short*)alloc((size_t)ROWS * D_MODEL * 2);
    float*          x1     = (float*)        alloc((size_t)ROWS * D_MODEL * 4);
    unsigned short* h2_b   = (unsigned short*)alloc((size_t)ROWS * D_MODEL * 2);
    unsigned short* g12_b  = (unsigned short*)alloc((size_t)ROWS * 2 * HIDDEN * 2);
    unsigned short* ffn_b  = (unsigned short*)alloc((size_t)ROWS * HIDDEN * 2);

    // 1) weight conversion to bf16 (QKV and W1|W2 column-concatenated)
    {
        int n;
        n = D_MODEL * D_MODEL;
        convert_concat<<<(n + 255) / 256, 256, 0, stream>>>(wq, wqkv_b, D_MODEL, 2048, 0, n);
        n = D_MODEL * (N_KV * HEAD_DIM);
        convert_concat<<<(n + 255) / 256, 256, 0, stream>>>(wk, wqkv_b, N_KV * HEAD_DIM, 2048, D_MODEL, n);
        convert_concat<<<(n + 255) / 256, 256, 0, stream>>>(wv, wqkv_b, N_KV * HEAD_DIM, 2048, D_MODEL + N_KV * HEAD_DIM, n);
        n = D_MODEL * D_MODEL;
        convert_concat<<<(n + 255) / 256, 256, 0, stream>>>(wo, wo_b, D_MODEL, D_MODEL, 0, n);
        n = D_MODEL * HIDDEN;
        convert_concat<<<(n + 255) / 256, 256, 0, stream>>>(w1, w12_b, HIDDEN, 2 * HIDDEN, 0, n);
        convert_concat<<<(n + 255) / 256, 256, 0, stream>>>(w2, w12_b, HIDDEN, 2 * HIDDEN, HIDDEN, n);
        n = HIDDEN * D_MODEL;
        convert_concat<<<(n + 255) / 256, 256, 0, stream>>>(w3, w3_b, D_MODEL, D_MODEL, 0, n);
    }

    // 2) h = rmsnorm(x, g1)  (bf16)
    rmsnorm_bf16<<<ROWS, 256, 0, stream>>>(x, g1, h_b);

    // 3) qkv = h @ [wq|wk|wv]   (8192 x 2048, K=1024)
    gemm_bf16_wmma<<<dim3(2048 / TN, ROWS / TM), 256, 0, stream>>>(
        h_b, wqkv_b, ROWS, 2048, D_MODEL, qkv_b, nullptr, nullptr);

    // 4) RoPE on q,k
    {
        int n = ROWS * (N_HEADS + N_KV) * (HEAD_DIM / 2);
        rope_qk<<<(n + 255) / 256, 256, 0, stream>>>(qkv_b);
    }

    // 5) causal GQA flash attention -> ctx (bf16, [b,t,h,d])
    flash_attn_wmma<<<dim3(SEQ / 64, BATCH * N_HEADS), 128, 0, stream>>>(qkv_b, ctx_b);

    // 6) x1 = x + ctx @ wo   (fp32)
    gemm_bf16_wmma<<<dim3(D_MODEL / TN, ROWS / TM), 256, 0, stream>>>(
        ctx_b, wo_b, ROWS, D_MODEL, D_MODEL, nullptr, x1, x);

    // 7) h2 = rmsnorm(x1, g2)
    rmsnorm_bf16<<<ROWS, 256, 0, stream>>>(x1, g2, h2_b);

    // 8) g12 = h2 @ [w1|w2]   (8192 x 8192, K=1024)
    gemm_bf16_wmma<<<dim3(2 * HIDDEN / TN, ROWS / TM), 256, 0, stream>>>(
        h2_b, w12_b, ROWS, 2 * HIDDEN, D_MODEL, g12_b, nullptr, nullptr);

    // 9) ffn_h = silu(g1part) * g2part
    {
        int n = ROWS * HIDDEN;
        silu_mul<<<(n + 255) / 256, 256, 0, stream>>>(g12_b, ffn_b);
    }

    // 10) out = x1 + ffn_h @ w3   (fp32, K=4096)
    gemm_bf16_wmma<<<dim3(D_MODEL / TN, ROWS / TM), 256, 0, stream>>>(
        ffn_b, w3_b, ROWS, D_MODEL, HIDDEN, nullptr, out, x1);
}
